// MIPNetwork_46316927320459
// MI455X (gfx1250) — matrix operand order repaired and checked by reading, not compiled, not run on hardware
//
#include <hip/hip_runtime.h>
#include <hip/hip_bf16.h>
#include <math.h>

typedef __attribute__((ext_vector_type(16))) __bf16 v16bf;
typedef __attribute__((ext_vector_type(8)))  __bf16 v8bf;
typedef __attribute__((ext_vector_type(8)))  float  v8f;

#define F_DIM 64
#define V_N 50000
#define C_N 50000
#define E_N 20000
#define NNZ_N 800000
#define NNZE_N 320000

// ---------------- elementwise / setup kernels ----------------

__global__ void k_fill(float* p, float v, int n) {
  int i = blockIdx.x * blockDim.x + threadIdx.x;
  if (i < n) p[i] = v;
}

__global__ void k_sq_scatter(const int* seg, const float* vals, float* out, int n) {
  int i = blockIdx.x * blockDim.x + threadIdx.x;
  if (i < n) atomicAdd(&out[seg[i]], vals[i] * vals[i]);
}

__global__ void k_scaler_fin(float* s, int n) {
  int i = blockIdx.x * blockDim.x + threadIdx.x;
  if (i < n) s[i] = sqrtf(s[i] + 1e-6f);
}

__global__ void k_sumsq(const float* x, int n, float* acc) {
  __shared__ float red[256];
  int i = blockIdx.x * 256 + threadIdx.x;
  float v = (i < n) ? x[i] * x[i] : 0.f;
  red[threadIdx.x] = v;
  __syncthreads();
  for (int s = 128; s > 0; s >>= 1) {
    if ((int)threadIdx.x < s) red[threadIdx.x] += red[threadIdx.x + s];
    __syncthreads();
  }
  if (threadIdx.x == 0) atomicAdd(acc, red[0]);
}

__global__ void k_obj_fin(const float* obj, const float* acc, float* out, int n) {
  int i = blockIdx.x * blockDim.x + threadIdx.x;
  if (i >= n) return;
  float denom = sqrtf(acc[0] / (float)n) + 1e-6f;
  out[i] = obj[i] / denom;
}

// Convert fp32 weight (K x N) into WMMA B-fragment-swizzled bf16 (Kp x Np padded).
// Fragment (kblk, ntile): lane holds 16 contiguous elements:
//   dst[(((kblk*(Np/16)) + ntile)*32 + lane)*16 + e]  <->  k = kblk*32 + (e>>3)*16 + (lane>>4)*8 + (e&7),
//                                                          n = ntile*16 + (lane&15)
__global__ void k_cvt_swz(const float* src, int K, int N, __bf16* dst, int Kp, int Np) {
  int idx = blockIdx.x * blockDim.x + threadIdx.x;
  if (idx >= Kp * Np) return;
  int e = idx & 15;
  int lane = (idx >> 4) & 31;
  int t = idx >> 9;
  int ntiles = Np >> 4;
  int ntile = t % ntiles;
  int kblk = t / ntiles;
  int k = kblk * 32 + (e >> 3) * 16 + (lane >> 4) * 8 + (e & 7);
  int n = ntile * 16 + (lane & 15);
  float v = (k < K && n < N) ? src[(size_t)k * N + n] : 0.f;
  dst[idx] = (__bf16)v;
}

__global__ void k_sigmoid(float* x, int n) {
  int i = blockIdx.x * blockDim.x + threadIdx.x;
  if (i < n) x[i] = 1.f / (1.f + expf(-x[i]));
}

// state = tmp[:, :F] + 0.5 * state   (tmp row stride ldt)
__global__ void k_residual(float* state, const float* tmp, int ldt, int rows) {
  int idx = blockIdx.x * blockDim.x + threadIdx.x;
  if (idx >= rows * F_DIM) return;
  int r = idx / F_DIM, f = idx - r * F_DIM;
  state[idx] = tmp[(size_t)r * ldt + f] + 0.5f * state[idx];
}

__global__ void k_rowdiv(float* x, const float* d, int rows) {
  int idx = blockIdx.x * blockDim.x + threadIdx.x;
  if (idx >= rows * F_DIM) return;
  x[idx] /= d[idx / F_DIM];
}

// ---------------- sparse segment-sum (SpMM) ----------------

__global__ void k_spmm(const int* seg, const int* gat, const float* vals, int nnz,
                       const float* X, int ldx, int ncols, float* out, int ldo, int coff) {
  int idx = blockIdx.x * blockDim.x + threadIdx.x;
  int e = idx / ncols;
  if (e >= nnz) return;
  int f = idx - e * ncols;
  float v = vals[e] * X[(size_t)gat[e] * ldx + f];
  atomicAdd(out + (size_t)seg[e] * ldo + coff + f, v);
}

// ---------------- concat staging (fp32 -> bf16, rows padded with zeros) -------

__global__ void k_stage(const float* s0, int w0, const float* s1, int w1,
                        const float* s2, int w2, const float* s3, int w3,
                        const float* s4, int w4, const float* s5, int w5,
                        int rows, int rowsPad, int Kp, __bf16* dst) {
  int idx = blockIdx.x * blockDim.x + threadIdx.x;
  int r = idx / Kp;
  if (r >= rowsPad) return;
  int o = idx - r * Kp;
  float v = 0.f;
  if (r < rows) {
    if (o < w0) v = s0[(size_t)r * w0 + o];
    else { o -= w0;
      if (o < w1) v = s1[(size_t)r * w1 + o];
      else { o -= w1;
        if (o < w2) v = s2[(size_t)r * w2 + o];
        else { o -= w2;
          if (o < w3) v = s3[(size_t)r * w3 + o];
          else { o -= w3;
            if (o < w4) v = s4[(size_t)r * w4 + o];
            else { o -= w4;
              if (o < w5) v = s5[(size_t)r * w5 + o];
    } } } } }
  }
  dst[idx] = (__bf16)v;
}

// ---------------- WMMA fragment helpers ----------------

__device__ __forceinline__ v16bf cat16(v8bf lo, v8bf hi) {
  return __builtin_shufflevector(lo, hi, 0, 1, 2, 3, 4, 5, 6, 7,
                                 8, 9, 10, 11, 12, 13, 14, 15);
}

// A 16x32 bf16 fragment: lane row = m0+(lane&15); elements 0..7 -> K=kb+hi*8+e,
// elements 8..15 -> K=kb+16+hi*8+(e-8).  Two contiguous 16B vector loads.
__device__ __forceinline__ v16bf frag_a(const __bf16* xrow, int kb, int hi) {
  v8bf lo = *(const v8bf*)(xrow + kb + hi * 8);
  v8bf hh = *(const v8bf*)(xrow + kb + 16 + hi * 8);
  return cat16(lo, hh);
}

// B fragment from swizzled weights: 32 contiguous bytes per lane.
__device__ __forceinline__ v16bf frag_b(const __bf16* Wb, int kblk, int ntiles,
                                        int nt, int lane) {
  const __bf16* wp = Wb + (((size_t)kblk * ntiles + nt) * 32 + lane) * 16;
  v8bf lo = *(const v8bf*)wp;
  v8bf hh = *(const v8bf*)(wp + 8);
  return cat16(lo, hh);
}

// ---------------- WMMA GEMM 1: H = X@W1 + b1, accumulate sum(H^2) ----------------
// block = 128 (4 waves); wave owns 16 rows x all 64 hidden cols (4 accumulators).
// grid.x = rowsPad/64.

__global__ void k_gemm1(const __bf16* __restrict__ Xb, int rows, int Kp,
                        const __bf16* __restrict__ Wb, const float* __restrict__ b1,
                        float* __restrict__ H, float* __restrict__ sumsq) {
  int lane = threadIdx.x & 31;
  int wave = threadIdx.x >> 5;
  int m0 = blockIdx.x * 64 + wave * 16;
  int hi = lane >> 4;
  int mrow = m0 + (lane & 15);
  const __bf16* xrow = Xb + (size_t)mrow * Kp;
  v8f acc[4] = {};
  for (int kb = 0; kb < Kp; kb += 32) {
    v16bf a = frag_a(xrow, kb, hi);
    int kblk = kb >> 5;
#pragma unroll
    for (int nt = 0; nt < 4; ++nt) {
      v16bf b = frag_b(Wb, kblk, 4, nt, lane);
      acc[nt] = __builtin_amdgcn_wmma_f32_16x16x32_bf16(false, a, false, b, (short)0,
                                                        acc[nt], false, false);
    }
  }
  float local = 0.f;
#pragma unroll
  for (int nt = 0; nt < 4; ++nt) {
    int n = nt * 16 + (lane & 15);
    float bias = b1[n];
#pragma unroll
    for (int r = 0; r < 8; ++r) {
      int m = m0 + r + hi * 8;
      float h = acc[nt][r] + bias;
      H[(size_t)m * 64 + n] = h;                 // padded rows written too (harmless)
      local += (m < rows) ? h * h : 0.f;         // branchless select
    }
  }
  __shared__ float red[128];
  red[threadIdx.x] = local;
  __syncthreads();
  for (int s = 64; s > 0; s >>= 1) {
    if ((int)threadIdx.x < s) red[threadIdx.x] += red[threadIdx.x + s];
    __syncthreads();
  }
  if (threadIdx.x == 0) atomicAdd(sumsq, red[0]);
}

// ---------------- WMMA GEMM 2: out = leaky(pairnorm(H)) @ W2 + b2 ----------------
// block = 128 (4 waves); wave owns 16 rows x NT*16 cols. grid.x = rowsPad/64.

template <int NT>
__global__ void k_gemm2(const float* __restrict__ H, int rows,
                        const __bf16* __restrict__ Wb,
                        const float* __restrict__ b2, int Dout,
                        const float* __restrict__ sumsq,
                        float* __restrict__ out, int ldo) {
  int lane = threadIdx.x & 31;
  int wave = threadIdx.x >> 5;
  int m0 = blockIdx.x * 64 + wave * 16;
  int hi = lane >> 4;
  int mrow = m0 + (lane & 15);
  float inv = rsqrtf(1e-6f + sumsq[0] / (float)rows);
  const float* hrow = H + (size_t)mrow * 64;
  v8f acc[NT] = {};
#pragma unroll
  for (int kb = 0; kb < 64; kb += 32) {
    v8f lo = *(const v8f*)(hrow + kb + hi * 8);
    v8f hh = *(const v8f*)(hrow + kb + 16 + hi * 8);
    v16bf a;
#pragma unroll
    for (int e = 0; e < 8; ++e) {
      float x = lo[e] * inv;
      x = (x > 0.f) ? x : 0.01f * x;
      a[e] = (__bf16)x;
      float y = hh[e] * inv;
      y = (y > 0.f) ? y : 0.01f * y;
      a[e + 8] = (__bf16)y;
    }
    int kblk = kb >> 5;
#pragma unroll
    for (int nt = 0; nt < NT; ++nt) {
      v16bf b = frag_b(Wb, kblk, NT, nt, lane);
      acc[nt] = __builtin_amdgcn_wmma_f32_16x16x32_bf16(false, a, false, b, (short)0,
                                                        acc[nt], false, false);
    }
  }
#pragma unroll
  for (int nt = 0; nt < NT; ++nt) {
    int n = nt * 16 + (lane & 15);
    if (n < Dout) {
      float bias = b2[n];
#pragma unroll
      for (int r = 0; r < 8; ++r) {
        int m = m0 + r + hi * 8;
        if (m < rows) out[(size_t)m * ldo + n] = acc[nt][r] + bias;
      }
    }
  }
}

// ---------------- output head: logistic noise + sigmoid mix ----------------

__global__ void k_output(const float* ov_in, const float* im, int step, float* dout, int n) {
  int i = blockIdx.x * blockDim.x + threadIdx.x;
  if (i >= n) return;
  float ov = ov_in[i];
  unsigned int h = (unsigned int)i * 0x9E3779B9u + (unsigned int)(step + 1) * 0x85EBCA6Bu;
  h ^= h >> 16; h *= 0x7feb352du; h ^= h >> 15; h *= 0x846ca68bu; h ^= h >> 16;
  float u = ((float)(h >> 8) + 0.5f) * (1.0f / 16777216.0f);
  const float eps = 1e-5f;
  u = eps + u * (1.f - 2.f * eps);
  float noise = logf(u / (1.f - u));
  float m = im[i];
  ov += noise * m;
  float sg = 1.f / (1.f + expf(-ov));
  dout[(size_t)step * n + i] = sg * m + ov * (1.f - m) * 0.1f;
  if (step == 2) dout[(size_t)3 * n + i] = ov;
}

// ---------------- host ----------------

extern "C" void kernel_launch(void* const* d_in, const int* in_sizes, int n_in,
                              void* d_out_v, int out_size, void* d_ws, size_t ws_size,
                              hipStream_t stream) {
  const int*   rows_i = (const int*)d_in[0];
  const int*   cols_i = (const int*)d_in[1];
  const float* vals_i = (const float*)d_in[2];
  const int*   rows_e = (const int*)d_in[3];
  const int*   cols_e = (const int*)d_in[4];
  const float* vals_e = (const float*)d_in[5];
  const float* cv     = (const float*)d_in[6];
  const float* ev     = (const float*)d_in[7];
  const float* rel    = (const float*)d_in[8];
  const float* obj    = (const float*)d_in[9];
  const float* im     = (const float*)d_in[10];
  // params flattened in sorted-key pytree order; each MLP: b1,b2,w1,w2
  const float* cu_b1 = (const float*)d_in[11];
  const float* cu_b2 = (const float*)d_in[12];
  const float* cu_w1 = (const float*)d_in[13];
  const float* cu_w2 = (const float*)d_in[14];
  const float* eq_b1 = (const float*)d_in[15];
  const float* eq_b2 = (const float*)d_in[16];
  const float* eq_w1 = (const float*)d_in[17];
  const float* eq_w2 = (const float*)d_in[18];
  const float* mq_b1 = (const float*)d_in[19];
  const float* mq_b2 = (const float*)d_in[20];
  const float* mq_w1 = (const float*)d_in[21];
  const float* mq_w2 = (const float*)d_in[22];
  const float* m2_b1 = (const float*)d_in[23];
  const float* m2_b2 = (const float*)d_in[24];
  const float* m2_w1 = (const float*)d_in[25];
  const float* m2_w2 = (const float*)d_in[26];
  const float* ou_b1 = (const float*)d_in[27];
  const float* ou_b2 = (const float*)d_in[28];
  const float* ou_w1 = (const float*)d_in[29];
  const float* ou_w2 = (const float*)d_in[30];
  const float* vu_b1 = (const float*)d_in[31];
  const float* vu_b2 = (const float*)d_in[32];
  const float* vu_w1 = (const float*)d_in[33];
  const float* vu_w2 = (const float*)d_in[34];
  const float* vg_b1 = (const float*)d_in[35];
  const float* vg_b2 = (const float*)d_in[36];
  const float* vg_w1 = (const float*)d_in[37];
  const float* vg_w2 = (const float*)d_in[38];
  (void)in_sizes; (void)n_in; (void)out_size;

  float* dout = (float*)d_out_v;

  // ---- workspace arena ----
  float* fp = (float*)d_ws;
  size_t o = 0;
  float* vars   = fp + o; o += (size_t)V_N * F_DIM;
  float* cons   = fp + o; o += (size_t)C_N * F_DIM;
  float* eqc    = fp + o; o += (size_t)E_N * F_DIM;
  float* scaler = fp + o; o += V_N;
  float* objn   = fp + o; o += V_N;
  float* bufA   = fp + o; o += (size_t)V_N * F_DIM;
  float* bufB   = fp + o; o += (size_t)V_N * F_DIM;
  float* bufC   = fp + o; o += (size_t)V_N * F_DIM;
  float* bufT   = fp + o; o += (size_t)C_N * 2 * F_DIM;
  float* bufH   = fp + o; o += (size_t)(V_N + 64) * F_DIM;   // padded rows
  float* ovb    = fp + o; o += V_N;
  float* scal   = fp + o; o += 16;
  __bf16* bp = (__bf16*)(fp + o);
  size_t ob = 0;
  __bf16* Xbf   = bp + ob; ob += (size_t)(V_N + 64) * 224;   // padded rows
  __bf16* W_cu1 = bp + ob; ob += 160 * 64;
  __bf16* W_cu2 = bp + ob; ob += 64 * 128;
  __bf16* W_eq1 = bp + ob; ob += 160 * 64;
  __bf16* W_eq2 = bp + ob; ob += 64 * 128;
  __bf16* W_mq1 = bp + ob; ob += 128 * 64;
  __bf16* W_mq2 = bp + ob; ob += 64 * 64;
  __bf16* W_m21 = bp + ob; ob += 128 * 64;
  __bf16* W_m22 = bp + ob; ob += 64 * 64;
  __bf16* W_ou1 = bp + ob; ob += 64 * 64;
  __bf16* W_ou2 = bp + ob; ob += 64 * 16;
  __bf16* W_vu1 = bp + ob; ob += 224 * 64;
  __bf16* W_vu2 = bp + ob; ob += 64 * 64;
  __bf16* W_vg1 = bp + ob; ob += 224 * 64;
  __bf16* W_vg2 = bp + ob; ob += 64 * 64;
  size_t need = o * sizeof(float) + ob * sizeof(__bf16);
  if (ws_size < need) return;  // insufficient scratch

  auto g1 = [](size_t n) { return dim3((unsigned)((n + 255) / 256)); };
  auto fill = [&](float* p, float v, size_t n) {
    k_fill<<<g1(n), 256, 0, stream>>>(p, v, (int)n);
  };
  auto cvt = [&](const float* s, int K, int N, __bf16* d, int Kp, int Np) {
    k_cvt_swz<<<g1((size_t)Kp * Np), 256, 0, stream>>>(s, K, N, d, Kp, Np);
  };
  auto spmm = [&](const int* seg, const int* gat, const float* vals, int nnz,
                  const float* X, int ldx, int ncols, float* out, int ldo, int coff) {
    k_spmm<<<g1((size_t)nnz * ncols), 256, 0, stream>>>(seg, gat, vals, nnz, X, ldx,
                                                        ncols, out, ldo, coff);
  };
  auto mlp = [&](const float* s0, int w0, const float* s1, int w1, const float* s2, int w2,
                 const float* s3, int w3, const float* s4, int w4, const float* s5, int w5,
                 int rows, int Kp, const __bf16* W1, const float* b1,
                 const __bf16* W2, int Np2, const float* b2, int Dout,
                 float* out, int ldo) {
    int rowsPad = (rows + 63) & ~63;
    k_fill<<<1, 1, 0, stream>>>(scal, 0.f, 1);
    k_stage<<<g1((size_t)rowsPad * Kp), 256, 0, stream>>>(
        s0, w0, s1, w1, s2, w2, s3, w3, s4, w4, s5, w5, rows, rowsPad, Kp, Xbf);
    k_gemm1<<<dim3(rowsPad / 64), 128, 0, stream>>>(Xbf, rows, Kp, W1, b1, bufH, scal);
    dim3 g2(rowsPad / 64);
    if (Np2 == 16)
      k_gemm2<1><<<g2, 128, 0, stream>>>(bufH, rows, W2, b2, Dout, scal, out, ldo);
    else if (Np2 == 64)
      k_gemm2<4><<<g2, 128, 0, stream>>>(bufH, rows, W2, b2, Dout, scal, out, ldo);
    else
      k_gemm2<8><<<g2, 128, 0, stream>>>(bufH, rows, W2, b2, Dout, scal, out, ldo);
  };

  // ---- setup ----
  fill(vars, 1.f, (size_t)V_N * F_DIM);
  fill(cons, 1.f, (size_t)C_N * F_DIM);
  fill(eqc, 1.f, (size_t)E_N * F_DIM);
  fill(scaler, 0.f, V_N);
  k_sq_scatter<<<g1(NNZ_N), 256, 0, stream>>>(rows_i, vals_i, scaler, NNZ_N);
  k_scaler_fin<<<g1(V_N), 256, 0, stream>>>(scaler, V_N);
  k_fill<<<1, 1, 0, stream>>>(scal + 1, 0.f, 1);
  k_sumsq<<<g1(V_N), 256, 0, stream>>>(obj, V_N, scal + 1);
  k_obj_fin<<<g1(V_N), 256, 0, stream>>>(obj, scal + 1, objn, V_N);

  cvt(cu_w1, 129, 64, W_cu1, 160, 64);  cvt(cu_w2, 64, 128, W_cu2, 64, 128);
  cvt(eq_w1, 129, 64, W_eq1, 160, 64);  cvt(eq_w2, 64, 128, W_eq2, 64, 128);
  cvt(mq_w1, 128, 64, W_mq1, 128, 64);  cvt(mq_w2, 64, 64, W_mq2, 64, 64);
  cvt(m2_w1, 128, 64, W_m21, 128, 64);  cvt(m2_w2, 64, 64, W_m22, 64, 64);
  cvt(ou_w1, 64, 64, W_ou1, 64, 64);    cvt(ou_w2, 64, 1, W_ou2, 64, 16);
  cvt(vu_w1, 195, 64, W_vu1, 224, 64);  cvt(vu_w2, 64, 64, W_vu2, 64, 64);
  cvt(vg_w1, 194, 64, W_vg1, 224, 64);  cvt(vg_w2, 64, 64, W_vg2, 64, 64);

  const float* Z = nullptr;

  // ---- 12 warmup steps + 3 output steps ----
  for (int s = 0; s < 15; ++s) {
    // c2g = spmm(rows_i, cols_i, vals_i, constraints, V)
    fill(bufA, 0.f, (size_t)V_N * F_DIM);
    spmm(rows_i, cols_i, vals_i, NNZ_N, cons, 64, 64, bufA, 64, 0);
    // c2ge = spmm(rows_e, cols_e, vals_e, eq_constraints, V)
    fill(bufB, 0.f, (size_t)V_N * F_DIM);
    spmm(rows_e, cols_e, vals_e, NNZE_N, eqc, 64, 64, bufB, 64, 0);
    // guess = sigmoid(mlp(variables_guess, [vars, c2g, c2ge, im, rel]))
    mlp(vars, 64, bufA, 64, bufB, 64, im, 1, rel, 1, Z, 0,
        V_N, 224, W_vg1, vg_b1, W_vg2, 64, vg_b2, 64, bufC, 64);
    k_sigmoid<<<g1((size_t)V_N * F_DIM), 256, 0, stream>>>(bufC, V_N * F_DIM);
    // vg-spmm to C: bufT = spmm(cols_i, rows_i, vals_i, [vars|guess], C)   (C x 128)
    fill(bufT, 0.f, (size_t)C_N * 128);
    spmm(cols_i, rows_i, vals_i, NNZ_N, vars, 64, 64, bufT, 128, 0);
    spmm(cols_i, rows_i, vals_i, NNZ_N, bufC, 64, 64, bufT, 128, 64);
    // v2m = mlp(make_query, bufT) -> bufA
    mlp(bufT, 128, Z, 0, Z, 0, Z, 0, Z, 0, Z, 0,
        C_N, 128, W_mq1, mq_b1, W_mq2, 64, mq_b2, 64, bufA, 64);
    // const_tmp = mlp(constraint_update, [cons, v2m, cv]) -> bufT (C x 128)
    mlp(cons, 64, bufA, 64, cv, 1, Z, 0, Z, 0, Z, 0,
        C_N, 160, W_cu1, cu_b1, W_cu2, 128, cu_b2, 128, bufT, 128);
    // constraints = const_tmp[:, :F] + 0.5*constraints
    k_residual<<<g1((size_t)C_N * F_DIM), 256, 0, stream>>>(cons, bufT, 128, C_N);
    // c2v = spmm(rows_i, cols_i, vals_i, const_tmp[:, F:], V) / vars_scaler -> bufB
    fill(bufB, 0.f, (size_t)V_N * F_DIM);
    spmm(rows_i, cols_i, vals_i, NNZ_N, bufT + 64, 128, 64, bufB, 64, 0);
    k_rowdiv<<<g1((size_t)V_N * F_DIM), 256, 0, stream>>>(bufB, scaler, V_N);
    // vg-spmm to E: bufT = spmm(cols_e, rows_e, vals_e, [vars|guess], E)   (E x 128)
    fill(bufT, 0.f, (size_t)E_N * 128);
    spmm(cols_e, rows_e, vals_e, NNZE_N, vars, 64, 64, bufT, 128, 0);
    spmm(cols_e, rows_e, vals_e, NNZE_N, bufC, 64, 64, bufT, 128, 64);
    // v2ce = mlp(make_query_2, bufT) -> bufA
    mlp(bufT, 128, Z, 0, Z, 0, Z, 0, Z, 0, Z, 0,
        E_N, 128, W_m21, m2_b1, W_m22, 64, m2_b2, 64, bufA, 64);
    // eq_tmp = mlp(eq_constraint_update, [eqc, v2ce, ev]) -> bufT (E x 128)
    mlp(eqc, 64, bufA, 64, ev, 1, Z, 0, Z, 0, Z, 0,
        E_N, 160, W_eq1, eq_b1, W_eq2, 128, eq_b2, 128, bufT, 128);
    // eq_constraints = eq_tmp[:, :F] + 0.5*eq_constraints
    k_residual<<<g1((size_t)E_N * F_DIM), 256, 0, stream>>>(eqc, bufT, 128, E_N);
    // eq2v = spmm(rows_e, cols_e, vals_e, eq_tmp[:, F:], V) -> bufC
    fill(bufC, 0.f, (size_t)V_N * F_DIM);
    spmm(rows_e, cols_e, vals_e, NNZE_N, bufT + 64, 128, 64, bufC, 64, 0);
    // vars = mlp(variable_update, [vars, eq2v, c2v, objn, im, rel]) + 0.5*vars
    mlp(vars, 64, bufC, 64, bufB, 64, objn, 1, im, 1, rel, 1,
        V_N, 224, W_vu1, vu_b1, W_vu2, 64, vu_b2, 64, bufA, 64);
    k_residual<<<g1((size_t)V_N * F_DIM), 256, 0, stream>>>(vars, bufA, 64, V_N);
    // output head on last 3 steps
    if (s >= 12) {
      mlp(vars, 64, Z, 0, Z, 0, Z, 0, Z, 0, Z, 0,
          V_N, 64, W_ou1, ou_b1, W_ou2, 16, ou_b2, 1, ovb, 1);
      k_output<<<g1(V_N), 256, 0, stream>>>(ovb, im, s - 12, dout, V_N);
    }
  }
}
